// Top1MoE_33621003993528
// MI455X (gfx1250) — matrix operand order repaired and checked by reading, not compiled, not run on hardware
//
#include <hip/hip_runtime.h>
#include <hip/hip_bf16.h>
#include <stdint.h>

#define D_MODEL 1024
#define HIDDEN  4096
#define N_EXP   8
#define N_TOK   8192   // B*S = 4*2048

typedef __attribute__((ext_vector_type(16))) __bf16 v16bf;
typedef __attribute__((ext_vector_type(8)))  float  v8f;

union ABFrag {
    v16bf v;
    uint4 q[2];
    unsigned short u[16];
};

__device__ __forceinline__ unsigned short f2bf(float f) {
    unsigned u = __float_as_uint(f);
    unsigned r = 0x7FFFu + ((u >> 16) & 1u);   // round-to-nearest-even
    return (unsigned short)((u + r) >> 16);
}

__device__ __forceinline__ float gelu_exact(float v) {
    return 0.5f * v * (1.0f + erff(v * 0.70710678118654752440f));
}

// ---------------- init ----------------
__global__ void init_counts_kernel(int* __restrict__ counts) {
    if (threadIdx.x < N_EXP) counts[threadIdx.x] = 0;
}

// ---------------- fp32 -> bf16 (straight) ----------------
__global__ void convert_x_kernel(const float* __restrict__ src,
                                 unsigned short* __restrict__ dst, int n4) {
    int i = blockIdx.x * blockDim.x + threadIdx.x;
    if (i >= n4) return;
    float4 f = ((const float4*)src)[i];
    ushort4 o;
    o.x = f2bf(f.x); o.y = f2bf(f.y); o.z = f2bf(f.z); o.w = f2bf(f.w);
    ((ushort4*)dst)[i] = o;
}

// ---------------- fp32 [e][K][N] -> bf16 [e][N][K] ----------------
__global__ void transpose_convert_kernel(const float* __restrict__ src,
                                         unsigned short* __restrict__ dst,
                                         int K, int N) {
    int e = blockIdx.y;
    size_t base = (size_t)e * K * N;
    int idx = blockIdx.x * blockDim.x + threadIdx.x;
    if (idx >= K * N) return;
    int k = idx / N;
    int n = idx - k * N;
    dst[base + (size_t)n * K + k] = f2bf(src[base + idx]);
}

// ---------------- router: one wave per token ----------------
__global__ __launch_bounds__(256) void router_kernel(
    const float* __restrict__ x, const float* __restrict__ Wr,
    const float* __restrict__ br, float* __restrict__ top_prob,
    int* __restrict__ counts, int* __restrict__ token_list) {
    int wave = threadIdx.x >> 5;
    int lane = threadIdx.x & 31;
    int tok  = blockIdx.x * 8 + wave;
    if (tok >= N_TOK) return;

    float acc[N_EXP];
#pragma unroll
    for (int e = 0; e < N_EXP; ++e) acc[e] = 0.0f;

    const float* xr = x + (size_t)tok * D_MODEL;
#pragma unroll 4
    for (int i = 0; i < D_MODEL / 32; ++i) {
        int d = lane + i * 32;
        float xv = xr[d];
        const float4* wr = (const float4*)(Wr + (size_t)d * N_EXP);
        float4 w0 = wr[0], w1 = wr[1];
        acc[0] += xv * w0.x; acc[1] += xv * w0.y;
        acc[2] += xv * w0.z; acc[3] += xv * w0.w;
        acc[4] += xv * w1.x; acc[5] += xv * w1.y;
        acc[6] += xv * w1.z; acc[7] += xv * w1.w;
    }
#pragma unroll
    for (int off = 16; off > 0; off >>= 1) {
#pragma unroll
        for (int e = 0; e < N_EXP; ++e)
            acc[e] += __shfl_xor(acc[e], off, 32);
    }
    if (lane == 0) {
        float l[N_EXP];
        float m = -3.4e38f;
        int idx = 0;
#pragma unroll
        for (int e = 0; e < N_EXP; ++e) {
            l[e] = acc[e] + br[e];
            if (l[e] > m) { m = l[e]; idx = e; }
        }
        float s = 0.0f;
#pragma unroll
        for (int e = 0; e < N_EXP; ++e) s += expf(l[e] - m);
        top_prob[tok] = 1.0f / s;
        int pos = atomicAdd(&counts[idx], 1);
        token_list[idx * N_TOK + pos] = tok;
    }
}

// ---------------- WMMA fragment loaders ----------------
// A: 16x32 bf16.  Lanes 0-15 = rows M=0..15 with K = k0+{0..15};
// lanes 16-31 = same rows with K = k0+{16..31}.
__device__ __forceinline__ v16bf load_a_lds(const unsigned short* lds,
                                            int stride, int k0, int lane) {
    ABFrag f;
    int row  = lane & 15;
    int base = k0 + ((lane & 16) ? 16 : 0);
    const uint4* p = (const uint4*)(lds + row * stride + base);
    f.q[0] = p[0];   // K base .. base+7
    f.q[1] = p[1];   // K base+8 .. base+15
    return f.v;
}
// B: 32x16 bf16.  Lane n<16 = column N=n, K=k0..k0+15 (16 contiguous in the
// pre-transposed [n][k] weight layout); lanes 16-31 = K=k0+16..k0+31.
__device__ __forceinline__ v16bf load_b_glb(const unsigned short* g) {
    ABFrag f;
    const uint4* p = (const uint4*)g;
    f.q[0] = p[0];
    f.q[1] = p[1];
    return f.v;
}

// ---------------- fused per-expert FFN: 16 tokens per block ----------------
__global__ __launch_bounds__(512) void moe_ffn_kernel(
    const unsigned short* __restrict__ xb,   // [tok][D] bf16
    const unsigned short* __restrict__ W1t,  // [e][H][D] bf16
    const unsigned short* __restrict__ W2t,  // [e][D][H] bf16
    const float* __restrict__ b1, const float* __restrict__ b2,
    const float* __restrict__ top_prob,
    const int* __restrict__ counts, const int* __restrict__ token_list,
    float* __restrict__ out) {

    __shared__ unsigned short lds_x[16 * D_MODEL];   // 32 KB
    __shared__ unsigned short lds_h[16 * HIDDEN];    // 128 KB
    __shared__ int   s_tok[16];
    __shared__ float s_prob[16];

    int e     = blockIdx.y;
    int mtile = blockIdx.x;
    int cnt   = counts[e];
    if (mtile * 16 >= cnt) return;

    int tid = threadIdx.x;
    if (tid < 16) {
        int row = mtile * 16 + tid;
        int rr  = (row < cnt) ? row : (mtile * 16);   // clamp to a valid row
        int t   = token_list[e * N_TOK + rr];
        s_tok[tid]  = t;
        s_prob[tid] = top_prob[t];
    }
    __syncthreads();

    // gather 16 token rows of bf16 x into LDS (64 B per thread)
    {
        int r = tid >> 5;
        int c = (tid & 31) * 32;
        const uint4* src = (const uint4*)(xb + (size_t)s_tok[r] * D_MODEL + c);
        uint4* dst = (uint4*)(lds_x + r * D_MODEL + c);
        dst[0] = src[0]; dst[1] = src[1]; dst[2] = src[2]; dst[3] = src[3];
    }
    __syncthreads();

    int w      = tid >> 5;          // wave id 0..15
    int lane   = tid & 31;
    int laneN  = lane & 15;
    int hiHalf = (lane >> 4) & 1;
    int bk     = hiHalf ? 16 : 0;

    // ---- GEMM1: H[16,4096] = X[16,1024] * W1 ; each wave owns 256 columns
    {
        v8f acc[16] = {};
        const unsigned short* w1e = W1t + (size_t)e * HIDDEN * D_MODEL;
        for (int k0 = 0; k0 < D_MODEL; k0 += 32) {
            v16bf a = load_a_lds(lds_x, D_MODEL, k0, lane);
#pragma unroll
            for (int nt = 0; nt < 16; ++nt) {
                int n = w * 256 + nt * 16 + laneN;
                v16bf b = load_b_glb(w1e + (size_t)n * D_MODEL + k0 + bk);
                acc[nt] = __builtin_amdgcn_wmma_f32_16x16x32_bf16(
                    false, a, false, b, (short)0, acc[nt], false, false);
            }
        }
        // epilogue: + b1, exact GELU, bf16 -> LDS
#pragma unroll
        for (int nt = 0; nt < 16; ++nt) {
            int col = w * 256 + nt * 16 + laneN;
            float bias = b1[e * HIDDEN + col];
#pragma unroll
            for (int r = 0; r < 8; ++r) {
                int m = r + hiHalf * 8;
                float v = acc[nt][r] + bias;
                lds_h[m * HIDDEN + col] = f2bf(gelu_exact(v));
            }
        }
    }
    __syncthreads();

    // ---- GEMM2: Y[16,1024] = H[16,4096] * W2 ; each wave owns 64 columns
    {
        v8f acc2[4] = {};
        const unsigned short* w2e = W2t + (size_t)e * D_MODEL * HIDDEN;
        for (int k0 = 0; k0 < HIDDEN; k0 += 32) {
            v16bf a = load_a_lds(lds_h, HIDDEN, k0, lane);
#pragma unroll
            for (int nt = 0; nt < 4; ++nt) {
                int n = w * 64 + nt * 16 + laneN;
                v16bf b = load_b_glb(w2e + (size_t)n * HIDDEN + k0 + bk);
                acc2[nt] = __builtin_amdgcn_wmma_f32_16x16x32_bf16(
                    false, a, false, b, (short)0, acc2[nt], false, false);
            }
        }
#pragma unroll
        for (int nt = 0; nt < 4; ++nt) {
            int col = w * 64 + nt * 16 + laneN;
            float bias = b2[e * D_MODEL + col];
#pragma unroll
            for (int r = 0; r < 8; ++r) {
                int m = r + hiHalf * 8;
                if (mtile * 16 + m < cnt) {
                    float v = acc2[nt][r] + bias;
                    out[(size_t)s_tok[m] * D_MODEL + col] = v * s_prob[m];
                }
            }
        }
    }
}

// ---------------- launch ----------------
extern "C" void kernel_launch(void* const* d_in, const int* in_sizes, int n_in,
                              void* d_out, int out_size, void* d_ws, size_t ws_size,
                              hipStream_t stream) {
    const float* x  = (const float*)d_in[0];
    const float* Wr = (const float*)d_in[1];
    const float* br = (const float*)d_in[2];
    const float* W1 = (const float*)d_in[3];
    const float* b1 = (const float*)d_in[4];
    const float* W2 = (const float*)d_in[5];
    const float* b2 = (const float*)d_in[6];
    float* out = (float*)d_out;

    char* ws = (char*)d_ws;
    size_t off = 0;
    int* counts = (int*)(ws + off);                 off += 256;
    int* token_list = (int*)(ws + off);             off += (size_t)N_EXP * N_TOK * sizeof(int);
    float* top_prob = (float*)(ws + off);           off += (size_t)N_TOK * sizeof(float);
    unsigned short* xb  = (unsigned short*)(ws + off); off += (size_t)N_TOK * D_MODEL * 2;
    unsigned short* W1t = (unsigned short*)(ws + off); off += (size_t)N_EXP * D_MODEL * HIDDEN * 2;
    unsigned short* W2t = (unsigned short*)(ws + off); off += (size_t)N_EXP * D_MODEL * HIDDEN * 2;

    init_counts_kernel<<<1, 32, 0, stream>>>(counts);

    int n4 = N_TOK * D_MODEL / 4;
    convert_x_kernel<<<(n4 + 255) / 256, 256, 0, stream>>>(x, xb, n4);

    dim3 tg((D_MODEL * HIDDEN + 255) / 256, N_EXP);
    transpose_convert_kernel<<<tg, 256, 0, stream>>>(W1, W1t, D_MODEL, HIDDEN);
    transpose_convert_kernel<<<tg, 256, 0, stream>>>(W2, W2t, HIDDEN, D_MODEL);

    router_kernel<<<N_TOK / 8, 256, 0, stream>>>(x, Wr, br, top_prob, counts, token_list);

    dim3 gg(N_TOK / 16, N_EXP);   // worst case: all tokens to one expert
    moe_ffn_kernel<<<gg, 512, 0, stream>>>(xb, W1t, W2t, b1, b2, top_prob,
                                           counts, token_list, out);
}
